// GraphNetBlock_31731218382878
// MI455X (gfx1250) — matrix operand order repaired and checked by reading, not compiled, not run on hardware
//
#include <hip/hip_runtime.h>

typedef __attribute__((ext_vector_type(16))) _Float16 v16h;
typedef __attribute__((ext_vector_type(8)))  _Float16 v8h;
typedef __attribute__((ext_vector_type(4)))  _Float16 v4h;
typedef __attribute__((ext_vector_type(8)))  float    v8f;

#define DD     128
#define TWO_D  256
#define TILE_M 64
#define BLOCK  256

// ---------------- prep kernels ----------------

__global__ void zero_f32_kernel(float* __restrict__ p, int n) {
  int i = (blockIdx.x * blockDim.x + threadIdx.x) * 4;
  if (i < n) *(float4*)(p + i) = make_float4(0.f, 0.f, 0.f, 0.f);
}

__global__ void cvt_nodes_kernel(const float* __restrict__ src, _Float16* __restrict__ dst, int n) {
  int i = (blockIdx.x * blockDim.x + threadIdx.x) * 4;
  if (i < n) {
    float4 f = *(const float4*)(src + i);
    v4h h; h[0] = (_Float16)f.x; h[1] = (_Float16)f.y; h[2] = (_Float16)f.z; h[3] = (_Float16)f.w;
    *(v4h*)(dst + i) = h;
  }
}

// W[k][n] (f32, row-major K x Ncols)  ->  Wt[n][k] (f16)
__global__ void cvt_wt_kernel(const float* __restrict__ W, _Float16* __restrict__ Wt,
                              int K, int Ncols) {
  int i = blockIdx.x * blockDim.x + threadIdx.x;
  if (i < K * Ncols) {
    int n = i / K, k = i - n * K;
    Wt[i] = (_Float16)W[(size_t)k * Ncols + n];
  }
}

// ---------------- WMMA fragment helpers ----------------

// B fragment (32x16 f16): lane col = lane&15, K = kb*32 + (lane&16?16:0) .. +15 contiguous
__device__ __forceinline__ v16h load_bfrag(const _Float16* __restrict__ wt, int K,
                                           int col, int kb, int lane) {
  int koff = kb * 32 + ((lane & 16) ? 16 : 0);
  return *(const v16h*)(wt + (size_t)col * K + koff);
}

// A fragment (16x32 f16) from LDS row-major [rows][stride]:
// lanes 0-15 : V0-7 = K 0..15 contiguous
// lanes 16-31: V0-7 = K 16..31 contiguous
__device__ __forceinline__ v16h load_afrag(const _Float16* s, int stride,
                                           int mrow, int kb, int lane) {
  int row  = mrow + (lane & 15);
  int base = kb * 32 + ((lane & 16) ? 16 : 0);
  v8h c0 = *(const v8h*)(s + row * stride + base);
  v8h c1 = *(const v8h*)(s + row * stride + base + 8);
  return __builtin_shufflevector(c0, c1, 0,1,2,3,4,5,6,7,8,9,10,11,12,13,14,15);
}

// One wave computes a 64x16 strip: 4 M-tiles of 16x16, K-loop of WMMAs.
template <int K>
__device__ __forceinline__ void wave_gemm(const _Float16* sIn, const _Float16* __restrict__ wt,
                                          int lane, int colbase, v8f acc[4]) {
#pragma unroll
  for (int kb = 0; kb < K / 32; ++kb) {
    v16h b = load_bfrag(wt, K, colbase + (lane & 15), kb, lane);
#pragma unroll
    for (int m = 0; m < 4; ++m) {
      v16h a = load_afrag(sIn, K, m * 16, kb, lane);
      acc[m] = __builtin_amdgcn_wmma_f32_16x16x32_f16(
          false, a, false, b, (short)0, acc[m], false, false);
    }
  }
}

// bias + ReLU + f16 restage to LDS [64][128] (D layout: row = v + 8*(lane>>4), col = lane&15)
__device__ __forceinline__ void relu_store(v8f acc[4], const float* __restrict__ bias,
                                           _Float16* sOut, int lane, int colbase) {
  int col   = colbase + (lane & 15);
  float bv  = bias[col];
  int rbase = (lane >> 4) * 8;
#pragma unroll
  for (int m = 0; m < 4; ++m)
#pragma unroll
    for (int v = 0; v < 8; ++v) {
      float h = acc[m][v] + bv;
      h = h > 0.f ? h : 0.f;
      sOut[(m * 16 + rbase + v) * DD + col] = (_Float16)h;
    }
}

__device__ __forceinline__ void zero_acc(v8f acc[4]) {
  v8f z = {};
#pragma unroll
  for (int m = 0; m < 4; ++m) acc[m] = z;
}

// ---------------- edge (message) kernel ----------------

__global__ __launch_bounds__(BLOCK) void edge_kernel(
    const _Float16* __restrict__ nodesH,
    const int* __restrict__ senders, const int* __restrict__ receivers,
    const _Float16* __restrict__ w1t, const float* __restrict__ b1,
    const _Float16* __restrict__ w2t, const float* __restrict__ b2,
    const _Float16* __restrict__ w3t, const float* __restrict__ b3,
    float* __restrict__ inbox, int E) {
  __shared__ __align__(16) _Float16 sA[TILE_M * TWO_D];
  __shared__ __align__(16) _Float16 sH[TILE_M * DD];
  __shared__ int sRecv[TILE_M];
  __shared__ int sSend[TILE_M];

  int tid = threadIdx.x;
  long e0 = (long)blockIdx.x * TILE_M;

  if (tid < TILE_M) {
    long e = e0 + tid;
    sRecv[tid] = (e < E) ? receivers[e] : 0;
  } else if (tid < 2 * TILE_M) {
    long e = e0 + (tid - TILE_M);
    sSend[tid - TILE_M] = (e < E) ? senders[e] : 0;
  }
  __syncthreads();

  // gather edge_in = [recv features | send features] -> sA [64][256] f16
#pragma unroll
  for (int i = 0; i < (TILE_M * 32) / BLOCK; ++i) {
    int c = tid + i * BLOCK;
    int row = c >> 5, sub = c & 31;
    int node = (sub < 16) ? sRecv[row] : sSend[row];
    int s = sub & 15;
    *(uint4*)(&sA[row * TWO_D + sub * 8]) =
        *(const uint4*)(nodesH + (size_t)node * DD + s * 8);
  }
  __syncthreads();

  int wave = tid >> 5, lane = tid & 31;
  int colbase = wave * 16;
  v8f acc[4];

  // layer 1: [64,256] x [256,128]
  zero_acc(acc);
  wave_gemm<TWO_D>(sA, w1t, lane, colbase, acc);
  relu_store(acc, b1, sH, lane, colbase);
  __syncthreads();

  // layer 2: [64,128] x [128,128]  (restage into sA reused as [64][128])
  zero_acc(acc);
  wave_gemm<DD>(sH, w2t, lane, colbase, acc);
  relu_store(acc, b2, sA, lane, colbase);
  __syncthreads();

  // layer 3 + scatter-add messages into inbox
  zero_acc(acc);
  wave_gemm<DD>(sA, w3t, lane, colbase, acc);
  int col   = colbase + (lane & 15);
  float bv  = b3[col];
  int rbase = (lane >> 4) * 8;
#pragma unroll
  for (int m = 0; m < 4; ++m)
#pragma unroll
    for (int v = 0; v < 8; ++v) {
      int row = m * 16 + rbase + v;
      if (e0 + row < E) {
        int recv = sRecv[row];
        atomicAdd(&inbox[(size_t)recv * DD + col], acc[m][v] + bv);
      }
    }
}

// ---------------- node-update kernel ----------------

__global__ __launch_bounds__(BLOCK) void node_kernel(
    const _Float16* __restrict__ nodesH, const float* __restrict__ nodesF,
    const _Float16* __restrict__ w1t, const float* __restrict__ b1,
    const _Float16* __restrict__ w2t, const float* __restrict__ b2,
    const _Float16* __restrict__ w3t, const float* __restrict__ b3,
    float* __restrict__ outInbox, int nNodes) {
  __shared__ __align__(16) _Float16 sA[TILE_M * TWO_D];
  __shared__ __align__(16) _Float16 sH[TILE_M * DD];

  int tid = threadIdx.x;
  int n0 = blockIdx.x * TILE_M;

  // gather node_in = [nodes f16 | inbox f32 -> f16] -> sA [64][256]
#pragma unroll
  for (int i = 0; i < (TILE_M * 32) / BLOCK; ++i) {
    int c = tid + i * BLOCK;
    int row = c >> 5, sub = c & 31;
    int n = n0 + row;
    if (n < nNodes) {
      if (sub < 16) {
        *(uint4*)(&sA[row * TWO_D + sub * 8]) =
            *(const uint4*)(nodesH + (size_t)n * DD + sub * 8);
      } else {
        const float4* f = (const float4*)(outInbox + (size_t)n * DD + (sub - 16) * 8);
        float4 f0 = f[0], f1 = f[1];
        v8h h;
        h[0] = (_Float16)f0.x; h[1] = (_Float16)f0.y; h[2] = (_Float16)f0.z; h[3] = (_Float16)f0.w;
        h[4] = (_Float16)f1.x; h[5] = (_Float16)f1.y; h[6] = (_Float16)f1.z; h[7] = (_Float16)f1.w;
        *(v8h*)(&sA[row * TWO_D + sub * 8]) = h;
      }
    } else {
      *(uint4*)(&sA[row * TWO_D + sub * 8]) = make_uint4(0, 0, 0, 0);
    }
  }
  __syncthreads();

  int wave = tid >> 5, lane = tid & 31;
  int colbase = wave * 16;
  v8f acc[4];

  zero_acc(acc);
  wave_gemm<TWO_D>(sA, w1t, lane, colbase, acc);
  relu_store(acc, b1, sH, lane, colbase);
  __syncthreads();

  zero_acc(acc);
  wave_gemm<DD>(sH, w2t, lane, colbase, acc);
  relu_store(acc, b2, sA, lane, colbase);
  __syncthreads();

  zero_acc(acc);
  wave_gemm<DD>(sA, w3t, lane, colbase, acc);
  int col   = colbase + (lane & 15);
  float bv  = b3[col];
  int rbase = (lane >> 4) * 8;
#pragma unroll
  for (int m = 0; m < 4; ++m)
#pragma unroll
    for (int v = 0; v < 8; ++v) {
      int row = m * 16 + rbase + v;
      int n = n0 + row;
      if (n < nNodes)  // residual: nodes + MLP(node_in); overwrites consumed inbox rows
        outInbox[(size_t)n * DD + col] = nodesF[(size_t)n * DD + col] + acc[m][v] + bv;
    }
}

// ---------------- launch ----------------

extern "C" void kernel_launch(void* const* d_in, const int* in_sizes, int n_in,
                              void* d_out, int out_size, void* d_ws, size_t ws_size,
                              hipStream_t stream) {
  const float* nodes     = (const float*)d_in[0];
  const int*   senders   = (const int*)d_in[1];
  const int*   receivers = (const int*)d_in[2];
  const float* mw1 = (const float*)d_in[3];  const float* mb1 = (const float*)d_in[4];
  const float* mw2 = (const float*)d_in[5];  const float* mb2 = (const float*)d_in[6];
  const float* mw3 = (const float*)d_in[7];  const float* mb3 = (const float*)d_in[8];
  const float* nw1 = (const float*)d_in[9];  const float* nb1 = (const float*)d_in[10];
  const float* nw2 = (const float*)d_in[11]; const float* nb2 = (const float*)d_in[12];
  const float* nw3 = (const float*)d_in[13]; const float* nb3 = (const float*)d_in[14];

  int nodeElems = in_sizes[0];        // N * 128
  int Nn = nodeElems / DD;
  int E  = in_sizes[1];
  float* out = (float*)d_out;         // doubles as the message inbox accumulator

  // workspace layout: nodes_f16, then six transposed f16 weight matrices
  char* ws = (char*)d_ws;
  _Float16* nodesH = (_Float16*)ws;
  size_t off = (((size_t)nodeElems * 2) + 255) & ~(size_t)255;
  _Float16* mw1t = (_Float16*)(ws + off); off += (size_t)TWO_D * DD * 2;
  _Float16* mw2t = (_Float16*)(ws + off); off += (size_t)DD * DD * 2;
  _Float16* mw3t = (_Float16*)(ws + off); off += (size_t)DD * DD * 2;
  _Float16* nw1t = (_Float16*)(ws + off); off += (size_t)TWO_D * DD * 2;
  _Float16* nw2t = (_Float16*)(ws + off); off += (size_t)DD * DD * 2;
  _Float16* nw3t = (_Float16*)(ws + off); off += (size_t)DD * DD * 2;
  (void)ws_size; (void)n_in; (void)out_size;

  int vecBlocks = (nodeElems / 4 + 255) / 256;
  zero_f32_kernel<<<vecBlocks, 256, 0, stream>>>(out, nodeElems);
  cvt_nodes_kernel<<<vecBlocks, 256, 0, stream>>>(nodes, nodesH, nodeElems);

  cvt_wt_kernel<<<(TWO_D * DD + 255) / 256, 256, 0, stream>>>(mw1, mw1t, TWO_D, DD);
  cvt_wt_kernel<<<(DD * DD + 255) / 256, 256, 0, stream>>>(mw2, mw2t, DD, DD);
  cvt_wt_kernel<<<(DD * DD + 255) / 256, 256, 0, stream>>>(mw3, mw3t, DD, DD);
  cvt_wt_kernel<<<(TWO_D * DD + 255) / 256, 256, 0, stream>>>(nw1, nw1t, TWO_D, DD);
  cvt_wt_kernel<<<(DD * DD + 255) / 256, 256, 0, stream>>>(nw2, nw2t, DD, DD);
  cvt_wt_kernel<<<(DD * DD + 255) / 256, 256, 0, stream>>>(nw3, nw3t, DD, DD);

  edge_kernel<<<(E + TILE_M - 1) / TILE_M, BLOCK, 0, stream>>>(
      nodesH, senders, receivers, mw1t, mb1, mw2t, mb2, mw3t, mb3, out, E);

  node_kernel<<<(Nn + TILE_M - 1) / TILE_M, BLOCK, 0, stream>>>(
      nodesH, nodes, nw1t, nb1, nw2t, nb2, nw3t, nb3, out, Nn);
}
